// LSTMCell_43181601194439
// MI455X (gfx1250) — matrix operand order repaired and checked by reading, not compiled, not run on hardware
//
#include <hip/hip_runtime.h>

// Fused LSTM cell for MI455X (gfx1250, wave32, WMMA).
// gates = [x|h] @ [Wi;Wh] + bias as one K=2048 GEMM on v_wmma_f32_16x16x32_bf16.
// Each workgroup accumulates all 4 gates for a 128x64 tile so the LSTM
// epilogue fuses in-register. Double-buffered LDS + register prefetch
// pipeline; packed fp32->bf16 conversion; wide (b128/b64) LDS staging.

#define BATCH 4096
#define HID   1024
#define FOURH 4096

typedef __bf16 bf16;
typedef __attribute__((ext_vector_type(4)))  float v4f;
typedef __attribute__((ext_vector_type(4)))  bf16  v4bf;
typedef __attribute__((ext_vector_type(8)))  bf16  v8bf;
typedef __attribute__((ext_vector_type(16))) bf16  v16bf;
typedef __attribute__((ext_vector_type(8)))  float v8f;

#define MT 128     // M tile (batch rows) per workgroup
#define NT 64      // N tile per gate per workgroup
#define KT 32      // K step
#define KTILES 64  // 2048 / 32
#define LDA 40     // A tile pitch (bf16): 80B rows, 16B-aligned chunks
#define LDB 72     // B tile pitch (bf16): 144B rows, 8B-aligned chunks

__device__ __forceinline__ v16bf ld_frag(const bf16* p0, const bf16* p1) {
  v8bf lo = *(const v8bf*)p0;               // ds_load_b128
  v8bf hi = *(const v8bf*)p1;               // ds_load_b128
  v16bf r;
#pragma unroll
  for (int i = 0; i < 8; ++i) { r[i] = lo[i]; r[i + 8] = hi[i]; }
  return r;
}

__device__ __forceinline__ float sigmoid_f(float x) {
  return 1.0f / (1.0f + __expf(-x));
}
__device__ __forceinline__ float tanh_f(float x) {
  float e = __expf(2.0f * x);
  return (e - 1.0f) / (e + 1.0f);
}

extern "C" __global__ __launch_bounds__(512)
void lstm_wmma_kernel(const float* __restrict__ x,
                      const float* __restrict__ h_tm1,
                      const float* __restrict__ c_tm1,
                      const float* __restrict__ wI,
                      const float* __restrict__ wH,
                      const float* __restrict__ bias,
                      float* __restrict__ h_out,
                      float* __restrict__ c_out) {
  __shared__ bf16 As[2][MT * LDA];        // double-buffered 128x32 act tile
  __shared__ bf16 Bs[2][4][KT * LDB];     // double-buffered per-gate 32x64 tiles

  const int tid    = threadIdx.x;
  const int lane   = tid & 31;
  const int wave   = tid >> 5;            // 0..15
  const int waveM  = wave >> 2;           // 0..3 -> 32-row slab
  const int waveN  = wave & 3;            // 0..3 -> 16-col slab
  const int laneM  = lane & 15;
  const int laneKg = lane >> 4;

  const int mBaseG = blockIdx.y * MT;
  const int nBaseG = blockIdx.x * NT;
  const int wm     = waveM * 32;
  const int wn     = waveN * 16;

  // staging coordinates: A in 8-elem chunks, B in 4-elem chunks per gate
  const int arow = tid >> 2, ac8 = (tid & 3) * 8;
  const int brow = tid >> 4, bc4 = (tid & 15) * 4;

  float biasv[4];
#pragma unroll
  for (int g = 0; g < 4; ++g)
    biasv[g] = bias[g * HID + nBaseG + wn + laneM];

  v8f acc[4][2] = {};                     // [gate][mfrag] f32 accumulators
  v4f pa0, pa1, pb[4];                    // register prefetch for next tile

  auto fetch = [&](int kt) {
    const float* Ag = (kt < 32) ? x  : h_tm1;      // concat [x|h] along K
    const float* Wg = (kt < 32) ? wI : wH;         // stack  [Wi;Wh] along K
    const int k0 = (kt & 31) * KT;
    const float* ap = &Ag[(mBaseG + arow) * HID + k0 + ac8];
    pa0 = *(const v4f*)ap;                          // global_load_b128
    pa1 = *(const v4f*)(ap + 4);
#pragma unroll
    for (int g = 0; g < 4; ++g)
      pb[g] = *(const v4f*)&Wg[(k0 + brow) * FOURH + g * HID + nBaseG + bc4];
  };

  auto commit = [&](int buf) {
    v4bf c0 = __builtin_convertvector(pa0, v4bf);  // packed f32->bf16
    v4bf c1 = __builtin_convertvector(pa1, v4bf);
    v8bf a8;
#pragma unroll
    for (int i = 0; i < 4; ++i) { a8[i] = c0[i]; a8[i + 4] = c1[i]; }
    *(v8bf*)&As[buf][arow * LDA + ac8] = a8;       // ds_store_b128
#pragma unroll
    for (int g = 0; g < 4; ++g)
      *(v4bf*)&Bs[buf][g][brow * LDB + bc4] =
          __builtin_convertvector(pb[g], v4bf);    // ds_store_b64
  };

  fetch(0);
  commit(0);

  for (int kt = 0; kt < KTILES; ++kt) {
    const int cur = kt & 1;
    __syncthreads();                               // buf[cur] staged & safe
    if (kt + 1 < KTILES) fetch(kt + 1);            // globals fly during WMMAs

    // A fragments: 16x32 bf16 (lane->M; elems 0..7 = K kg*8.., 8..15 = 16+kg*8..)
    v16bf afrag[2];
#pragma unroll
    for (int mf = 0; mf < 2; ++mf) {
      const bf16* base = &As[cur][(wm + mf * 16 + laneM) * LDA + laneKg * 8];
      afrag[mf] = ld_frag(base, base + 16);
    }
    // B fragments per gate: 32x16 bf16 (lane->N; elems 0..7 = K kg*8.., 8..15 = 16+kg*8..)
#pragma unroll
    for (int g = 0; g < 4; ++g) {
      const bf16* bbase = &Bs[cur][g][(laneKg * 8) * LDB + wn + laneM];
      v16bf bfrag;
#pragma unroll
      for (int i = 0; i < 8; ++i) {
        bfrag[i] = bbase[i * LDB];
        bfrag[i + 8] = bbase[(16 + i) * LDB];
      }
#pragma unroll
      for (int mf = 0; mf < 2; ++mf) {
        acc[g][mf] = __builtin_amdgcn_wmma_f32_16x16x32_bf16(
            false, afrag[mf], false, bfrag, (short)0, acc[g][mf],
            false, false);
      }
    }
    if (kt + 1 < KTILES) commit(cur ^ 1);          // fill the other buffer
  }

  // fused LSTM epilogue: C/D layout lane->N (lane&15), VGPR r->M (+8 if lane>=16)
  const int nGl = nBaseG + wn + laneM;             // gate-local column 0..1023
#pragma unroll
  for (int mf = 0; mf < 2; ++mf) {
#pragma unroll
    for (int r = 0; r < 8; ++r) {
      int m = mBaseG + wm + mf * 16 + (laneKg << 3) + r;
      float f_t  = sigmoid_f(acc[0][mf][r] + biasv[0]);
      float i_t  = sigmoid_f(acc[1][mf][r] + biasv[1]);
      float cand = tanh_f   (acc[2][mf][r] + biasv[2]);
      float o_t  = sigmoid_f(acc[3][mf][r] + biasv[3]);
      float cprev = __builtin_nontemporal_load(&c_tm1[m * HID + nGl]);
      float ct = f_t * cprev + i_t * cand;
      float ht = o_t * tanh_f(ct);
      __builtin_nontemporal_store(ht, &h_out[m * HID + nGl]);
      __builtin_nontemporal_store(ct, &c_out[m * HID + nGl]);
    }
  }
}

extern "C" void kernel_launch(void* const* d_in, const int* in_sizes, int n_in,
                              void* d_out, int out_size, void* d_ws, size_t ws_size,
                              hipStream_t stream) {
  const float* x    = (const float*)d_in[0];
  const float* h    = (const float*)d_in[1];
  const float* c    = (const float*)d_in[2];
  const float* wI   = (const float*)d_in[3];
  const float* wH   = (const float*)d_in[4];
  const float* bias = (const float*)d_in[5];
  float* h_out = (float*)d_out;
  float* c_out = h_out + (size_t)BATCH * HID;      // outputs concatenated (h_t, c_t)

  dim3 grid(HID / NT, BATCH / MT);                 // 16 x 32 workgroups
  lstm_wmma_kernel<<<grid, 512, 0, stream>>>(x, h, c, wI, wH, bias, h_out, c_out);
}
